// LSTMTagger_40896678592548
// MI455X (gfx1250) — matrix operand order repaired and checked by reading, not compiled, not run on hardware
//
#include <hip/hip_runtime.h>
#include <math.h>

// ---------------------------------------------------------------------------
// LSTM tagger for MI455X (gfx1250, wave32, WMMA f32_16x16x32_f16)
//
// Pipeline:
//  K0  convert weights f32->f16 into ws; gather word embeddings into Zbuf
//  K1  char LSTM: 32 WGs x 64 words, weights resident in LDS, WMMA gates,
//      masked ragged-length state update, writes h_char into Zbuf[:,256:384]
//  K2  Xw = Zbuf @ Wih_w^T + b_w   (2048x384 x 384x1024, WMMA)
//  K3  word LSTM: single persistent 1024-thread WG, 2048 sequential steps,
//      thread j owns gate row j, h broadcast via LDS, Whh_w streamed as f16,
//      next-step gate inputs prefetched (global_prefetch_b8)
//  K4  logits = H @ W_out^T + b_out, fused log_softmax (WMMA + LDS reduce)
// ---------------------------------------------------------------------------

typedef __attribute__((ext_vector_type(16))) _Float16 v16h;
typedef __attribute__((ext_vector_type(8)))  _Float16 v8h;
typedef __attribute__((ext_vector_type(8)))  float    v8f;

union V16U { v16h v; v8h h[2]; };

// A fragment (16x32 f16, M x K), row-major source [row][k], per ISA 7.12.2:
// lanes 0-15: K=0..7 and 16..23 of row M=lane; lanes 16-31: K=8..15, 24..31.
static __device__ __forceinline__ v16h frag_a(const _Float16* base, int rowBase,
                                              int stride, int kBase, int l16, int hi) {
  const _Float16* p = base + (rowBase + l16) * stride + kBase + hi * 8;
  V16U u;
  u.h[0] = *(const v8h*)p;
  u.h[1] = *(const v8h*)(p + 16);
  return u.v;
}

// B fragment (32x16 f16, K x N), weights stored row-major [n][k]:
// lanes 0-15: K=0..7 and 16..23 of column N=lane%16; lanes 16-31: K=8..15, 24..31.
static __device__ __forceinline__ v16h frag_b(const _Float16* base, int colBase,
                                              int stride, int kBase, int l16, int hi) {
  const _Float16* p = base + (colBase + l16) * stride + kBase + hi * 8;
  V16U u;
  u.h[0] = *(const v8h*)p;
  u.h[1] = *(const v8h*)(p + 16);
  return u.v;
}

static __device__ __forceinline__ v8f wmma16(v16h a, v16h b, v8f c) {
  return __builtin_amdgcn_wmma_f32_16x16x32_f16(false, a, false, b, (short)0, c,
                                                false, false);
}

static __device__ __forceinline__ float sigf(float x) {
  return 1.0f / (1.0f + __expf(-x));
}

// ---------------------------------------------------------------------------
// K0a: generic f32 -> f16 convert
__global__ void cvt_f32_f16_kernel(const float* __restrict__ src,
                                   _Float16* __restrict__ dst, int n) {
  for (int i = blockIdx.x * blockDim.x + threadIdx.x; i < n;
       i += gridDim.x * blockDim.x)
    dst[i] = (_Float16)src[i];
}

// K0b: gather word embeddings into Zbuf[word][0:256] (f16), row stride 384
__global__ void gather_emb_kernel(const int* __restrict__ sentence,
                                  const float* __restrict__ W_emb,
                                  _Float16* __restrict__ Zbuf) {
  int i = blockIdx.x * blockDim.x + threadIdx.x;  // [2048*256)
  int w = i >> 8, d = i & 255;
  int tok = sentence[w];
  Zbuf[w * 384 + d] = (_Float16)W_emb[tok * 256 + d];
}

// ---------------------------------------------------------------------------
// K1: char LSTM. 64 words per block, 256 threads (8 waves).
// Wave w: mTile = w>>1 (16-word row tile), hcHalf = w&1 (64 hidden cols).
__global__ __launch_bounds__(256) void char_lstm_kernel(
    const int* __restrict__ chars, const int* __restrict__ char_len,
    const _Float16* __restrict__ WihC16, const _Float16* __restrict__ WhhC16,
    const float* __restrict__ b_c, const float* __restrict__ C_emb,
    _Float16* __restrict__ Zbuf) {
  __shared__ _Float16 sWih[512 * 64];       //  64 KB
  __shared__ _Float16 sWhh[512 * 128];      // 128 KB
  __shared__ _Float16 sH[2][64 * 128];      //  32 KB ping-pong hidden state
  __shared__ _Float16 sX[64 * 64];          //   8 KB char embeddings at step t

  const int tid = threadIdx.x;
  const int wordBase = blockIdx.x * 64;

  for (int i = tid * 8; i < 512 * 64; i += 2048)
    *(v8h*)(sWih + i) = *(const v8h*)(WihC16 + i);
  for (int i = tid * 8; i < 512 * 128; i += 2048)
    *(v8h*)(sWhh + i) = *(const v8h*)(WhhC16 + i);
  for (int i = tid * 8; i < 64 * 128; i += 2048) {
    v8h z = {};
    *(v8h*)(sH[0] + i) = z;
  }

  const int lane = tid & 31, wave = tid >> 5;
  const int l16 = lane & 15, hi = lane >> 4;
  const int mTile = wave >> 1, hcHalf = wave & 1;
  const int mBase = mTile * 16;

  int lenr[8];
#pragma unroll
  for (int r = 0; r < 8; r++) {
    int L = char_len[wordBase + mBase + r + 8 * hi];
    lenr[r] = L < 1 ? 1 : L;
  }

  float hs[4][8], cs[4][8];
#pragma unroll
  for (int q = 0; q < 4; q++)
#pragma unroll
    for (int r = 0; r < 8; r++) {
      hs[q][r] = 0.0f;
      cs[q][r] = 0.0f;
    }

  __syncthreads();

  for (int t = 0; t < 16; t++) {
    // gather x_t = C_emb[chars[:, t]] -> sX (f16). One word-quarter per thread.
    {
      int w = tid >> 2, seg = (tid & 3) * 16;
      int ci = chars[(wordBase + w) * 16 + t];
      const float* ce = C_emb + ci * 64 + seg;
      _Float16* dst = sX + w * 64 + seg;
#pragma unroll
      for (int e = 0; e < 16; e++) dst[e] = (_Float16)ce[e];
    }
    __syncthreads();

    const _Float16* hRead = sH[t & 1];
    _Float16* hWrite = sH[(t + 1) & 1];

#pragma unroll
    for (int q = 0; q < 4; q++) {
      const int hcT = hcHalf * 4 + q;
      const int nI = hcT * 16, nF = 128 + hcT * 16, nG = 256 + hcT * 16,
                nO = 384 + hcT * 16;
      v8f aI, aF, aG, aO;
      {
        float bI = b_c[nI + l16], bF = b_c[nF + l16], bG = b_c[nG + l16],
              bO = b_c[nO + l16];
#pragma unroll
        for (int r = 0; r < 8; r++) {
          aI[r] = bI; aF[r] = bF; aG[r] = bG; aO[r] = bO;
        }
      }
      // x @ Wih_c^T  (K = 64)
#pragma unroll
      for (int kc = 0; kc < 2; kc++) {
        v16h A = frag_a(sX, mBase, 64, kc * 32, l16, hi);
        aI = wmma16(A, frag_b(sWih, nI, 64, kc * 32, l16, hi), aI);
        aF = wmma16(A, frag_b(sWih, nF, 64, kc * 32, l16, hi), aF);
        aG = wmma16(A, frag_b(sWih, nG, 64, kc * 32, l16, hi), aG);
        aO = wmma16(A, frag_b(sWih, nO, 64, kc * 32, l16, hi), aO);
      }
      // h @ Whh_c^T  (K = 128)
#pragma unroll
      for (int kc = 0; kc < 4; kc++) {
        v16h A = frag_a(hRead, mBase, 128, kc * 32, l16, hi);
        aI = wmma16(A, frag_b(sWhh, nI, 128, kc * 32, l16, hi), aI);
        aF = wmma16(A, frag_b(sWhh, nF, 128, kc * 32, l16, hi), aF);
        aG = wmma16(A, frag_b(sWhh, nG, 128, kc * 32, l16, hi), aG);
        aO = wmma16(A, frag_b(sWhh, nO, 128, kc * 32, l16, hi), aO);
      }
      // masked LSTM cell update (freeze past word length)
#pragma unroll
      for (int r = 0; r < 8; r++) {
        float ig = sigf(aI[r]), fg = sigf(aF[r]);
        float gg = tanhf(aG[r]), og = sigf(aO[r]);
        float cn = fg * cs[q][r] + ig * gg;
        float hn = og * tanhf(cn);
        bool valid = t < lenr[r];
        cs[q][r] = valid ? cn : cs[q][r];
        hs[q][r] = valid ? hn : hs[q][r];
        hWrite[(mBase + r + 8 * hi) * 128 + hcT * 16 + l16] = (_Float16)hs[q][r];
      }
    }
    __syncthreads();
  }

  // final char hidden -> Zbuf[word][256:384]
#pragma unroll
  for (int q = 0; q < 4; q++)
#pragma unroll
    for (int r = 0; r < 8; r++) {
      int w = wordBase + mBase + r + 8 * hi;
      Zbuf[w * 384 + 256 + (hcHalf * 4 + q) * 16 + l16] = (_Float16)hs[q][r];
    }
}

// ---------------------------------------------------------------------------
// K2: Xw = Zbuf @ Wih_w^T + b_w.  M=2048 N=1024 K=384.
// Block: 64 rows x 128 gate cols, 256 threads (8 waves).
__global__ __launch_bounds__(256) void gemm_xw_kernel(
    const _Float16* __restrict__ Zbuf, const _Float16* __restrict__ WihW16,
    const float* __restrict__ b_w, float* __restrict__ Xw) {
  __shared__ _Float16 sZ[64 * 384];    // 48 KB
  __shared__ _Float16 sW[128 * 384];   // 96 KB

  const int tid = threadIdx.x;
  const int bm = blockIdx.x & 31, bn = blockIdx.x >> 5;
  const _Float16* zsrc = Zbuf + bm * 64 * 384;
  const _Float16* wsrc = WihW16 + bn * 128 * 384;
  for (int i = tid * 8; i < 64 * 384; i += 2048)
    *(v8h*)(sZ + i) = *(const v8h*)(zsrc + i);
  for (int i = tid * 8; i < 128 * 384; i += 2048)
    *(v8h*)(sW + i) = *(const v8h*)(wsrc + i);
  __syncthreads();

  const int lane = tid & 31, wave = tid >> 5;
  const int l16 = lane & 15, hi = lane >> 4;
  const int mTile = wave >> 1, nGroup = wave & 1;
  const int mBase = mTile * 16;

  v8f acc[4];
#pragma unroll
  for (int nt = 0; nt < 4; nt++) {
    float bv = b_w[bn * 128 + (nGroup * 4 + nt) * 16 + l16];
#pragma unroll
    for (int r = 0; r < 8; r++) acc[nt][r] = bv;
  }

#pragma unroll 4
  for (int kc = 0; kc < 12; kc++) {
    v16h A = frag_a(sZ, mBase, 384, kc * 32, l16, hi);
#pragma unroll
    for (int nt = 0; nt < 4; nt++) {
      v16h B = frag_b(sW, (nGroup * 4 + nt) * 16, 384, kc * 32, l16, hi);
      acc[nt] = wmma16(A, B, acc[nt]);
    }
  }

#pragma unroll
  for (int nt = 0; nt < 4; nt++)
#pragma unroll
    for (int r = 0; r < 8; r++) {
      int gm = bm * 64 + mBase + r + 8 * hi;
      int gn = bn * 128 + (nGroup * 4 + nt) * 16 + l16;
      Xw[gm * 1024 + gn] = acc[nt][r];
    }
}

// ---------------------------------------------------------------------------
// K3: word LSTM. Single 1024-thread workgroup, 2048 sequential steps.
// Thread j owns gate row j (i/f/g/o interleaved by 256). Whh_w rows are f16
// and stay L2-resident (512 KB); h is broadcast through LDS. The next step's
// gate-input row is prefetched (global_prefetch_b8) to shorten the serial
// critical path.
__global__ __launch_bounds__(1024) void word_lstm_kernel(
    const float* __restrict__ Xw, const _Float16* __restrict__ WhhW16,
    _Float16* __restrict__ Hout16) {
  __shared__ float hCur[256];
  __shared__ float gbuf[1024];

  const int j = threadIdx.x;
  float c = 0.0f;
  if (j < 256) hCur[j] = 0.0f;
  const _Float16* wrow = WhhW16 + j * 256;
  __builtin_prefetch(&Xw[j], 0, 0);
  __syncthreads();

  for (int t = 0; t < 2048; t++) {
    float acc = Xw[t * 1024 + j];  // includes x@Wih_w^T + b_w
    {
      int tn = t + 1 < 2048 ? t + 1 : t;
      __builtin_prefetch(&Xw[tn * 1024 + j], 0, 0);  // hide L2 latency of t+1
    }
#pragma unroll 8
    for (int kk = 0; kk < 256; kk += 8) {
      union { float4 f; _Float16 h[8]; } u;
      u.f = *(const float4*)(wrow + kk);
#pragma unroll
      for (int e = 0; e < 8; e++) acc += (float)u.h[e] * hCur[kk + e];
    }
    gbuf[j] = acc;
    __syncthreads();
    if (j < 256) {
      float ig = sigf(gbuf[j]);
      float fg = sigf(gbuf[256 + j]);
      float gg = tanhf(gbuf[512 + j]);
      float og = sigf(gbuf[768 + j]);
      c = fg * c + ig * gg;
      float h = og * tanhf(c);
      hCur[j] = h;
      Hout16[t * 256 + j] = (_Float16)h;
    }
    __syncthreads();
  }
}

// ---------------------------------------------------------------------------
// K4: logits = H @ W_out^T + b_out, fused log_softmax. 64 rows/block, 4 waves.
__global__ __launch_bounds__(128) void logits_kernel(
    const _Float16* __restrict__ Hout16, const _Float16* __restrict__ WOut16,
    const float* __restrict__ b_out, float* __restrict__ out) {
  __shared__ _Float16 sHm[64 * 256];  // 32 KB
  __shared__ _Float16 sWm[64 * 256];  // 32 KB
  __shared__ float sL[64 * 64];       // 16 KB

  const int tid = threadIdx.x;
  const _Float16* hsrc = Hout16 + blockIdx.x * 64 * 256;
  for (int i = tid * 8; i < 64 * 256; i += 1024) {
    *(v8h*)(sHm + i) = *(const v8h*)(hsrc + i);
    *(v8h*)(sWm + i) = *(const v8h*)(WOut16 + i);
  }
  __syncthreads();

  const int lane = tid & 31, wave = tid >> 5;
  const int l16 = lane & 15, hi = lane >> 4;
  const int mBase = wave * 16;

  v8f acc[4];
#pragma unroll
  for (int nt = 0; nt < 4; nt++) {
    float bv = b_out[nt * 16 + l16];
#pragma unroll
    for (int r = 0; r < 8; r++) acc[nt][r] = bv;
  }
#pragma unroll
  for (int kc = 0; kc < 8; kc++) {
    v16h A = frag_a(sHm, mBase, 256, kc * 32, l16, hi);
#pragma unroll
    for (int nt = 0; nt < 4; nt++)
      acc[nt] = wmma16(A, frag_b(sWm, nt * 16, 256, kc * 32, l16, hi), acc[nt]);
  }
#pragma unroll
  for (int nt = 0; nt < 4; nt++)
#pragma unroll
    for (int r = 0; r < 8; r++)
      sL[(mBase + r + 8 * hi) * 64 + nt * 16 + l16] = acc[nt][r];
  __syncthreads();

  if (tid < 64) {
    const float* row = sL + tid * 64;
    float m = -1e30f;
    for (int cidx = 0; cidx < 64; cidx++) m = fmaxf(m, row[cidx]);
    float s = 0.0f;
    for (int cidx = 0; cidx < 64; cidx++) s += __expf(row[cidx] - m);
    float ls = m + __logf(s);
    float* orow = out + (blockIdx.x * 64 + tid) * 64;
    for (int cidx = 0; cidx < 64; cidx++) orow[cidx] = row[cidx] - ls;
  }
}

// ---------------------------------------------------------------------------
extern "C" void kernel_launch(void* const* d_in, const int* in_sizes, int n_in,
                              void* d_out, int out_size, void* d_ws,
                              size_t ws_size, hipStream_t stream) {
  const int*   sentence = (const int*)d_in[0];
  const int*   chars    = (const int*)d_in[1];
  const int*   char_len = (const int*)d_in[2];
  const float* W_emb    = (const float*)d_in[3];
  const float* C_emb    = (const float*)d_in[4];
  const float* Wih_c    = (const float*)d_in[5];
  const float* Whh_c    = (const float*)d_in[6];
  const float* b_c      = (const float*)d_in[7];
  const float* Wih_w    = (const float*)d_in[8];
  const float* Whh_w    = (const float*)d_in[9];
  const float* b_w      = (const float*)d_in[10];
  const float* W_out    = (const float*)d_in[11];
  const float* b_out    = (const float*)d_in[12];
  float* out = (float*)d_out;

  char* ws = (char*)d_ws;
  size_t off = 0;
  auto take = [&](size_t bytes) {
    size_t o = off;
    off = (off + bytes + 255) & ~(size_t)255;
    return o;
  };
  _Float16* Zbuf   = (_Float16*)(ws + take((size_t)2048 * 384 * 2));
  _Float16* WihC16 = (_Float16*)(ws + take((size_t)512 * 64 * 2));
  _Float16* WhhC16 = (_Float16*)(ws + take((size_t)512 * 128 * 2));
  _Float16* WihW16 = (_Float16*)(ws + take((size_t)1024 * 384 * 2));
  _Float16* WhhW16 = (_Float16*)(ws + take((size_t)1024 * 256 * 2));
  _Float16* WOut16 = (_Float16*)(ws + take((size_t)64 * 256 * 2));
  float*    Xw     = (float*)(ws + take((size_t)2048 * 1024 * 4));
  _Float16* Hout16 = (_Float16*)(ws + take((size_t)2048 * 256 * 2));
  (void)in_sizes; (void)n_in; (void)out_size; (void)ws_size;

  // K0: weight conversion + embedding gather
  cvt_f32_f16_kernel<<<64, 256, 0, stream>>>(Wih_c, WihC16, 512 * 64);
  cvt_f32_f16_kernel<<<128, 256, 0, stream>>>(Whh_c, WhhC16, 512 * 128);
  cvt_f32_f16_kernel<<<512, 256, 0, stream>>>(Wih_w, WihW16, 1024 * 384);
  cvt_f32_f16_kernel<<<512, 256, 0, stream>>>(Whh_w, WhhW16, 1024 * 256);
  cvt_f32_f16_kernel<<<64, 256, 0, stream>>>(W_out, WOut16, 64 * 256);
  gather_emb_kernel<<<2048, 256, 0, stream>>>(sentence, W_emb, Zbuf);

  // K1: char LSTM (WMMA gates, LDS-resident weights)
  char_lstm_kernel<<<32, 256, 0, stream>>>(chars, char_len, WihC16, WhhC16,
                                           b_c, C_emb, Zbuf);
  // K2: word-LSTM input projection (WMMA GEMM)
  gemm_xw_kernel<<<256, 256, 0, stream>>>(Zbuf, WihW16, b_w, Xw);
  // K3: sequential word LSTM (persistent single workgroup)
  word_lstm_kernel<<<1, 1024, 0, stream>>>(Xw, WhhW16, Hout16);
  // K4: logits + log_softmax (WMMA GEMM, fused reduce)
  logits_kernel<<<32, 128, 0, stream>>>(Hout16, WOut16, b_out, out);
}
